// FlowMatchingShield_20100446945404
// MI455X (gfx1250) — matrix operand order repaired and checked by reading, not compiled, not run on hardware
//
#include <hip/hip_runtime.h>
#include <hip/hip_bf16.h>

typedef __bf16 bf16_t;
typedef __attribute__((ext_vector_type(16))) __bf16 v16bf;
typedef __attribute__((ext_vector_type(8)))  float  v8f;

#define HID   1024
#define BATCH 1024
#define NSTEPS 50

// ---------------------------------------------------------------------------
// CDNA5 async global->LDS copy (16B per lane), tracked by ASYNCcnt.
// lds_off: LDS byte address (low 32 bits of the generic shared pointer);
// gptr:    per-lane global address.
// ---------------------------------------------------------------------------
__device__ __forceinline__ void async_copy_b128(unsigned int lds_off, const void* gptr) {
    asm volatile("global_load_async_to_lds_b128 %0, %1, off"
                 :: "v"(lds_off), "v"((unsigned long long)(uintptr_t)gptr)
                 : "memory");
}

__device__ __forceinline__ void wait_asynccnt0() {
#if __has_builtin(__builtin_amdgcn_s_wait_asynccnt)
    __builtin_amdgcn_s_wait_asynccnt(0);
#else
    asm volatile("s_wait_asynccnt 0" ::: "memory");
#endif
}

// ---------------------------------------------------------------------------
// init x = x0  (d_out is poisoned before timing; we own its initialization)
// ---------------------------------------------------------------------------
__global__ void init_x_kernel(const float* __restrict__ x0, float* __restrict__ x) {
    int idx = blockIdx.x * blockDim.x + threadIdx.x;
    if (idx < BATCH * 2) x[idx] = x0[idx];
}

// ---------------------------------------------------------------------------
// Wt[n][k] = bf16(W[k][n])  (one-time; row stride of W is N)
// ---------------------------------------------------------------------------
__global__ void cast_transpose_kernel(const float* __restrict__ W,
                                      bf16_t* __restrict__ Wt, int K, int N) {
    int idx = blockIdx.x * blockDim.x + threadIdx.x;
    if (idx >= K * N) return;
    int n = idx / K;
    int k = idx - n * K;
    Wt[idx] = (bf16_t)W[k * N + n];
}

// ---------------------------------------------------------------------------
// spad[r][k] = bf16(state[r][k]) for k<254, else 0   (K padded to 256)
// ---------------------------------------------------------------------------
__global__ void build_spad_kernel(const float* __restrict__ state,
                                  bf16_t* __restrict__ spad) {
    int idx = blockIdx.x * blockDim.x + threadIdx.x;   // 1024*256
    int r = idx >> 8;
    int k = idx & 255;
    spad[idx] = (k < 254) ? (bf16_t)state[r * 254 + k] : (bf16_t)0.0f;
}

// ---------------------------------------------------------------------------
// stepb[i][c] = temb(t_i) @ W0[256:320, :]   (temb identical across batch)
// ---------------------------------------------------------------------------
__global__ void build_stepb_kernel(const float* __restrict__ W0,
                                   float* __restrict__ stepb) {
    int idx = blockIdx.x * blockDim.x + threadIdx.x;   // 50*1024
    if (idx >= NSTEPS * HID) return;
    int i = idx >> 10;
    int c = idx & 1023;
    float t = (float)i * 0.02f;                        // dt = 1/50
    float acc = 0.0f;
    #pragma unroll 8
    for (int j = 0; j < 32; ++j) {
        float f   = __expf((float)j * (-9.210340371976184f / 31.0f)); // ln(1e4)
        float ang = t * f;
        acc += sinf(ang) * W0[(256 + j)      * HID + c];
        acc += cosf(ang) * W0[(256 + 32 + j) * HID + c];
    }
    stepb[idx] = acc;
}

// ---------------------------------------------------------------------------
// h1 = silu(pre0 + x @ W0[254:256,:] + stepb_i)   (rank-2 update, bf16 out)
// ---------------------------------------------------------------------------
__global__ void layer0_kernel(const float* __restrict__ pre0,
                              const float* __restrict__ x,
                              const float* __restrict__ W0,
                              const float* __restrict__ stepb_i,
                              bf16_t* __restrict__ h1) {
    int idx = blockIdx.x * blockDim.x + threadIdx.x;   // 1024*256 threads, 4 cols each
    int r  = idx >> 8;
    int c4 = (idx & 255) * 4;
    float x0v = x[r * 2 + 0];
    float x1v = x[r * 2 + 1];
    const float4 p  = *(const float4*)&pre0[r * HID + c4];
    const float4 w0 = *(const float4*)&W0[254 * HID + c4];
    const float4 w1 = *(const float4*)&W0[255 * HID + c4];
    const float4 sb = *(const float4*)&stepb_i[c4];
    union { bf16_t o[4]; uint2 u; } out;
    float h;
    h = p.x + x0v * w0.x + x1v * w1.x + sb.x; out.o[0] = (bf16_t)(h / (1.0f + __expf(-h)));
    h = p.y + x0v * w0.y + x1v * w1.y + sb.y; out.o[1] = (bf16_t)(h / (1.0f + __expf(-h)));
    h = p.z + x0v * w0.z + x1v * w1.z + sb.z; out.o[2] = (bf16_t)(h / (1.0f + __expf(-h)));
    h = p.w + x0v * w0.w + x1v * w1.w + sb.w; out.o[3] = (bf16_t)(h / (1.0f + __expf(-h)));
    *(uint2*)&h1[r * HID + c4] = out.u;
}

// ---------------------------------------------------------------------------
// WMMA GEMM:  C[M][N] = act(A[M][K] @ Bt[N][K]^T + bias)
// bf16 inputs, fp32 accumulation via v_wmma_f32_16x16x32_bf16.
// Block tile 128x128, K-tile 32, 8 wave32s; each wave does a 64x32 tile.
// Double-buffered LDS filled with global_load_async_to_lds_b128 (ASYNCcnt).
// ---------------------------------------------------------------------------
template <bool SILU, typename OutT>
__global__ __launch_bounds__(256) void wmma_gemm_kernel(
        const bf16_t* __restrict__ A,   // [M][K] row-major
        const bf16_t* __restrict__ Bt,  // [N][K] row-major (B transposed)
        const float*  __restrict__ bias,
        OutT* __restrict__ C,           // [M][N]
        int M, int N, int K) {
    constexpr int BM = 128, BN = 128, BK = 32;
    __shared__ bf16_t sA[2][BM * BK];   // 2 x 8 KB
    __shared__ bf16_t sB[2][BN * BK];   // 2 x 8 KB

    const int tid  = threadIdx.x;
    const int m0   = blockIdx.y * BM;
    const int n0   = blockIdx.x * BN;
    const int wid  = tid >> 5;
    const int lane = tid & 31;
    const int half = lane >> 4;      // 0: lanes 0-15, 1: lanes 16-31
    const int mr   = lane & 15;
    const int wm   = (wid & 1) * 64; // wave M offset within block
    const int wn   = (wid >> 1) * 32;

    // staging assignment: thread covers 2 x 16B chunks of each tile
    const int srow0 = tid >> 2;                 // 0..63
    const int srow1 = (tid + 256) >> 2;         // 64..127
    const int sseg  = (tid & 3) * 8;            // element offset 0,8,16,24
    const unsigned int sA0 = (unsigned int)(uintptr_t)&sA[0][0];
    const unsigned int sB0 = (unsigned int)(uintptr_t)&sB[0][0];

    v8f acc[4][2];
    #pragma unroll
    for (int i = 0; i < 4; ++i)
        #pragma unroll
        for (int j = 0; j < 2; ++j)
            acc[i][j] = v8f{0.f, 0.f, 0.f, 0.f, 0.f, 0.f, 0.f, 0.f};

    auto stage_async = [&](int kt, int buf) {
        const int k0 = kt * BK;
        const unsigned int bo = (unsigned int)buf * (BM * BK * 2);  // bytes
        async_copy_b128(sA0 + bo + (unsigned int)(srow0 * BK + sseg) * 2,
                        &A[(size_t)(m0 + srow0) * K + k0 + sseg]);
        async_copy_b128(sB0 + bo + (unsigned int)(srow0 * BK + sseg) * 2,
                        &Bt[(size_t)(n0 + srow0) * K + k0 + sseg]);
        async_copy_b128(sA0 + bo + (unsigned int)(srow1 * BK + sseg) * 2,
                        &A[(size_t)(m0 + srow1) * K + k0 + sseg]);
        async_copy_b128(sB0 + bo + (unsigned int)(srow1 * BK + sseg) * 2,
                        &Bt[(size_t)(n0 + srow1) * K + k0 + sseg]);
    };

    const int ktiles = K / BK;
    stage_async(0, 0);
    wait_asynccnt0();
    __syncthreads();

    for (int kt = 0; kt < ktiles; ++kt) {
        const int buf = kt & 1;
        if (kt + 1 < ktiles) stage_async(kt + 1, (kt + 1) & 1);  // overlap with compute

        // A frag (16x32): lane half 0 holds K {0..7,16..23}, half 1 holds {8..15,24..31}
        // B frag (32x16): lane (N=mr) holds K {0..7,16..23} (half 0) / {8..15,24..31} (half 1)
        union Frag { v16bf v; uint4 u[2]; };
        Frag a[4], b[2];
        #pragma unroll
        for (int j = 0; j < 2; ++j) {
            int col = wn + j * 16 + mr;
            b[j].u[0] = *(const uint4*)&sB[buf][col * BK + half * 8];
            b[j].u[1] = *(const uint4*)&sB[buf][col * BK + half * 8 + 16];
        }
        #pragma unroll
        for (int i = 0; i < 4; ++i) {
            int row  = wm + i * 16 + mr;
            int base = half * 8;
            a[i].u[0] = *(const uint4*)&sA[buf][row * BK + base];
            a[i].u[1] = *(const uint4*)&sA[buf][row * BK + base + 16];
        }
        #pragma unroll
        for (int i = 0; i < 4; ++i)
            #pragma unroll
            for (int j = 0; j < 2; ++j)
                acc[i][j] = __builtin_amdgcn_wmma_f32_16x16x32_bf16(
                    false, a[i].v, false, b[j].v, (short)0, acc[i][j], false, false);

        wait_asynccnt0();     // next tile landed in the other buffer
        __syncthreads();      // all waves done reading this buffer
    }

    // epilogue: C/D layout — VGPR r: lanes 0-15 -> M=r, lanes 16-31 -> M=r+8
    #pragma unroll
    for (int j = 0; j < 2; ++j) {
        int col  = n0 + wn + j * 16 + mr;
        float bv = bias[col];
        #pragma unroll
        for (int i = 0; i < 4; ++i) {
            #pragma unroll
            for (int r = 0; r < 8; ++r) {
                int row = m0 + wm + i * 16 + half * 8 + r;
                float v = acc[i][j][r] + bv;
                if (SILU) v = v / (1.0f + __expf(-v));
                C[(size_t)row * N + col] = (OutT)v;
            }
        }
    }
}

// ---------------------------------------------------------------------------
// Final layer (N=2) + Euler update:  x += dt * (h4 @ W4 + b4)
// one wave32 per row; shuffle reduction
// ---------------------------------------------------------------------------
__global__ void final_kernel(const bf16_t* __restrict__ h4,
                             const float* __restrict__ W4,   // [1024][2]
                             const float* __restrict__ b4,   // [2]
                             float* __restrict__ x, float dt) {
    int wid  = (blockIdx.x << 3) + (threadIdx.x >> 5);  // row 0..1023
    int lane = threadIdx.x & 31;
    float acc0 = 0.f, acc1 = 0.f;
    for (int k = lane; k < HID; k += 32) {
        float hv = (float)h4[(size_t)wid * HID + k];
        acc0 += hv * W4[2 * k + 0];
        acc1 += hv * W4[2 * k + 1];
    }
    #pragma unroll
    for (int off = 16; off > 0; off >>= 1) {
        acc0 += __shfl_xor(acc0, off, 32);
        acc1 += __shfl_xor(acc1, off, 32);
    }
    if (lane == 0) {
        x[2 * wid + 0] += dt * (acc0 + b4[0]);
        x[2 * wid + 1] += dt * (acc1 + b4[1]);
    }
}

// ---------------------------------------------------------------------------
extern "C" void kernel_launch(void* const* d_in, const int* in_sizes, int n_in,
                              void* d_out, int out_size, void* d_ws, size_t ws_size,
                              hipStream_t stream) {
    (void)in_sizes; (void)n_in; (void)out_size; (void)ws_size;
    const float* state = (const float*)d_in[0];
    const float* x0    = (const float*)d_in[1];
    const float* W0 = (const float*)d_in[2];
    const float* b0 = (const float*)d_in[3];
    const float* W1 = (const float*)d_in[4];
    const float* b1 = (const float*)d_in[5];
    const float* W2 = (const float*)d_in[6];
    const float* b2 = (const float*)d_in[7];
    const float* W3 = (const float*)d_in[8];
    const float* b3 = (const float*)d_in[9];
    const float* W4 = (const float*)d_in[10];
    const float* b4 = (const float*)d_in[11];
    float* x = (float*)d_out;

    char* ws = (char*)d_ws;
    size_t off = 0;
    float*  pre0  = (float*)(ws + off);  off += (size_t)HID * HID * 4;   // 4 MB
    bf16_t* hA    = (bf16_t*)(ws + off); off += (size_t)HID * HID * 2;   // 2 MB
    bf16_t* hB    = (bf16_t*)(ws + off); off += (size_t)HID * HID * 2;   // 2 MB
    bf16_t* Wt1   = (bf16_t*)(ws + off); off += (size_t)HID * HID * 2;   // 2 MB
    bf16_t* Wt2   = (bf16_t*)(ws + off); off += (size_t)HID * HID * 2;   // 2 MB
    bf16_t* Wt3   = (bf16_t*)(ws + off); off += (size_t)HID * HID * 2;   // 2 MB
    bf16_t* Wt0p  = (bf16_t*)(ws + off); off += (size_t)HID * 256 * 2;   // 512 KB
    bf16_t* spad  = (bf16_t*)(ws + off); off += (size_t)HID * 256 * 2;   // 512 KB
    float*  stepb = (float*)(ws + off);  off += (size_t)NSTEPS * HID * 4;

    dim3 blk(256);
    dim3 g88(8, 8);

    // one-time prep (graph-captured; deterministic)
    init_x_kernel<<<8, blk, 0, stream>>>(x0, x);
    cast_transpose_kernel<<<4096, blk, 0, stream>>>(W1, Wt1, HID, HID);
    cast_transpose_kernel<<<4096, blk, 0, stream>>>(W2, Wt2, HID, HID);
    cast_transpose_kernel<<<4096, blk, 0, stream>>>(W3, Wt3, HID, HID);
    cast_transpose_kernel<<<1024, blk, 0, stream>>>(W0, Wt0p, 256, HID);  // rows 0..255
    build_spad_kernel<<<1024, blk, 0, stream>>>(state, spad);
    build_stepb_kernel<<<200, blk, 0, stream>>>(W0, stepb);
    // pre0 = [state,0,0] @ W0[0:256,:] + b0  (step-invariant part of layer 0)
    wmma_gemm_kernel<false, float><<<g88, blk, 0, stream>>>(spad, Wt0p, b0, pre0,
                                                            HID, HID, 256);

    for (int i = 0; i < NSTEPS; ++i) {
        layer0_kernel<<<1024, blk, 0, stream>>>(pre0, x, W0, stepb + (size_t)i * HID, hA);
        wmma_gemm_kernel<true, bf16_t><<<g88, blk, 0, stream>>>(hA, Wt1, b1, hB, HID, HID, HID);
        wmma_gemm_kernel<true, bf16_t><<<g88, blk, 0, stream>>>(hB, Wt2, b2, hA, HID, HID, HID);
        wmma_gemm_kernel<true, bf16_t><<<g88, blk, 0, stream>>>(hA, Wt3, b3, hB, HID, HID, HID);
        final_kernel<<<128, blk, 0, stream>>>(hB, W4, b4, x, 0.02f);
    }
}
